// GCN_55155970015429
// MI455X (gfx1250) — matrix operand order repaired and checked by reading, not compiled, not run on hardware
//
#include <hip/hip_runtime.h>

// ---------- WMMA vector types (gfx1250, wave32) ----------
typedef __attribute__((ext_vector_type(16))) __bf16          v16bf;
typedef __attribute__((ext_vector_type(8)))  float           v8f;
typedef __attribute__((ext_vector_type(8)))  unsigned short  v8us;
typedef __attribute__((ext_vector_type(16))) unsigned short  v16us;

static __device__ __forceinline__ unsigned short f32_to_bf16(float f) {
  unsigned int u = __float_as_uint(f);
  u += 0x7FFFu + ((u >> 16) & 1u);           // round-to-nearest-even
  return (unsigned short)(u >> 16);
}

// ---------- elementwise helpers ----------
__global__ void k_fill(float* __restrict__ p, long long count, float v) {
  long long i = (long long)blockIdx.x * blockDim.x + threadIdx.x;
  if (i < count) p[i] = v;
}

__global__ void k_deg(const int* __restrict__ col, long long E, float* __restrict__ deg) {
  long long e = (long long)blockIdx.x * blockDim.x + threadIdx.x;
  if (e < E) atomicAdd(&deg[col[e]], 1.0f);
}

__global__ void k_rsqrt_inplace(float* __restrict__ d, long long n) {
  long long i = (long long)blockIdx.x * blockDim.x + threadIdx.x;
  if (i < n) d[i] = rsqrtf(d[i]);            // deg >= 1 (self loop) so always finite
}

__global__ void k_cvt_bf16(const float* __restrict__ src, unsigned short* __restrict__ dst,
                           long long count) {
  long long i = (long long)blockIdx.x * blockDim.x + threadIdx.x;
  if (i < count) dst[i] = f32_to_bf16(src[i]);
}

// W[K x Nin] (row-major) -> Wt[Npad x K] bf16, zero-padded columns
__global__ void k_cvt_transpose_bf16(const float* __restrict__ W, unsigned short* __restrict__ Wt,
                                     int K, int Nin, int Npad) {
  long long idx = (long long)blockIdx.x * blockDim.x + threadIdx.x;
  long long total = (long long)Npad * K;
  if (idx >= total) return;
  int nn = (int)(idx / K);
  int kk = (int)(idx % K);
  float v = (nn < Nin) ? W[(long long)kk * Nin + nn] : 0.0f;
  Wt[idx] = f32_to_bf16(v);
}

// agg -> bf16( relu(agg + b[f]) )
__global__ void k_bias_relu_cvt(const float* __restrict__ agg, const float* __restrict__ b,
                                unsigned short* __restrict__ out, long long count, int F) {
  long long i = (long long)blockIdx.x * blockDim.x + threadIdx.x;
  if (i >= count) return;
  int f = (int)(i % F);
  float v = agg[i] + b[f];
  out[i] = f32_to_bf16(v > 0.0f ? v : 0.0f);
}

// out = relu(out + b[f])  (in place, f32)
__global__ void k_bias_relu_f32(float* __restrict__ out, const float* __restrict__ b,
                                long long count, int F) {
  long long i = (long long)blockIdx.x * blockDim.x + threadIdx.x;
  if (i >= count) return;
  int f = (int)(i % F);
  float v = out[i] + b[f];
  out[i] = v > 0.0f ? v : 0.0f;
}

// ---------- bf16 WMMA GEMM:  C[MxN] = A[MxK] * Bt[NxK]^T  (f32 accumulate) ----------
// One wave computes a 16x64 strip. grid = (ceil(M/16), N/256), block = 128 (4 waves).
// Requires: K % 32 == 0, N % 256 == 0.
__global__ __launch_bounds__(128) void k_gemm_bf16(const unsigned short* __restrict__ A,
                                                   const unsigned short* __restrict__ Bt,
                                                   float* __restrict__ C,
                                                   int M, int K, int N) {
  const int lane  = threadIdx.x & 31;
  const int wave  = threadIdx.x >> 5;
  const int mTile = blockIdx.x;                    // 16 rows
  const int n0    = blockIdx.y * 256 + wave * 64;  // 64 cols per wave
  const int mloc  = lane & 15;
  const int half  = lane >> 4;
  int row = mTile * 16 + mloc;
  if (row >= M) row = M - 1;                       // clamp loads (stores guarded)
  const unsigned short* Arow  = A  + (long long)row * K;
  const unsigned short* Bbase = Bt + (long long)(n0 + mloc) * K;  // column n0+nloc of B

  v8f acc0 = {}, acc1 = {}, acc2 = {}, acc3 = {};

  for (int k0 = 0; k0 < K; k0 += 32) {
    // A fragment: lane half h holds K = 16h .. 16h+15 contiguous (matches B)
    v16us au  = *(const v16us*)(Arow + k0 + 16 * half);
    v16bf a   = __builtin_bit_cast(v16bf, au);

    // B fragment (contiguous k order, same as A): lane holds column n, K = 16*half .. 16*half+15 contiguous
    const long long bk = (long long)(k0 + 16 * half);
    v16us b0u = *(const v16us*)(Bbase + 0LL * 16 * K + bk);
    v16us b1u = *(const v16us*)(Bbase + 1LL * 16 * K + bk);
    v16us b2u = *(const v16us*)(Bbase + 2LL * 16 * K + bk);
    v16us b3u = *(const v16us*)(Bbase + 3LL * 16 * K + bk);

    acc0 = __builtin_amdgcn_wmma_f32_16x16x32_bf16(false, a, false, __builtin_bit_cast(v16bf, b0u),
                                                   (short)0, acc0, false, false);
    acc1 = __builtin_amdgcn_wmma_f32_16x16x32_bf16(false, a, false, __builtin_bit_cast(v16bf, b1u),
                                                   (short)0, acc1, false, false);
    acc2 = __builtin_amdgcn_wmma_f32_16x16x32_bf16(false, a, false, __builtin_bit_cast(v16bf, b2u),
                                                   (short)0, acc2, false, false);
    acc3 = __builtin_amdgcn_wmma_f32_16x16x32_bf16(false, a, false, __builtin_bit_cast(v16bf, b3u),
                                                   (short)0, acc3, false, false);
  }

  // C/D layout: VGPR r, lane half h -> row m = r + 8h ; col = n0 + 16t + (lane&15)
#pragma unroll
  for (int r = 0; r < 8; ++r) {
    int m = mTile * 16 + r + 8 * half;
    if (m < M) {
      long long base = (long long)m * N + n0 + mloc;
      C[base +  0] = acc0[r];
      C[base + 16] = acc1[r];
      C[base + 32] = acc2[r];
      C[base + 48] = acc3[r];
    }
  }
}

// ---------- edge aggregation: out[c] += H[r] * dinv[r]*dinv[c]  (+ self loops) ----------
// One wave per edge; lanes split the feature dimension. Edges e>=E are self-loops (r=c=e-E).
__global__ __launch_bounds__(256) void k_aggregate(const float* __restrict__ H,
                                                   const int* __restrict__ row,
                                                   const int* __restrict__ col,
                                                   const float* __restrict__ dinv,
                                                   long long E, int n, int F, int ldh,
                                                   float* __restrict__ out) {
  long long e = (long long)blockIdx.x * 8 + (threadIdx.x >> 5);
  long long total = E + n;
  if (e >= total) return;
  int lane = threadIdx.x & 31;
  int r, c;
  if (e < E) { r = row[e]; c = col[e]; }
  else       { r = c = (int)(e - E); }
  float nrm = dinv[r] * dinv[c];
  const float* hr = H + (long long)r * ldh;
  float* oc = out + (long long)c * F;
  for (int f = lane * 4; f < F; f += 128) {
    if (f + 3 < F) {
      float4 v = *(const float4*)(hr + f);
      atomicAdd(oc + f + 0, v.x * nrm);
      atomicAdd(oc + f + 1, v.y * nrm);
      atomicAdd(oc + f + 2, v.z * nrm);
      atomicAdd(oc + f + 3, v.w * nrm);
    } else {
      for (int j = f; j < F; ++j) atomicAdd(oc + j, hr[j] * nrm);
    }
  }
}

// ---------- launcher ----------
extern "C" void kernel_launch(void* const* d_in, const int* in_sizes, int n_in,
                              void* d_out, int out_size, void* d_ws, size_t ws_size,
                              hipStream_t stream) {
  const float* x    = (const float*)d_in[0];
  const int*   ei   = (const int*)d_in[1];
  const float* W1   = (const float*)d_in[2];
  const float* b1   = (const float*)d_in[3];
  const float* W2   = (const float*)d_in[4];
  const float* b2   = (const float*)d_in[5];
  float* out = (float*)d_out;

  const int hid   = in_sizes[3];                 // 512
  const int f_out = in_sizes[5];                 // 250
  const int f_in  = in_sizes[2] / hid;           // 256
  const int n     = in_sizes[0] / f_in;          // 50000
  const long long E = in_sizes[1] / 2;           // 1.6M
  const int f_out_pad = ((f_out + 255) / 256) * 256;  // 256

  const int* row = ei;
  const int* col = ei + E;

  // workspace layout (256B aligned slots)
  char* ws = (char*)d_ws;
  size_t off = 0;
  auto alloc = [&](size_t bytes) { size_t o = off; off = (off + bytes + 255) & ~(size_t)255; return o; };
  size_t o_dinv = alloc((size_t)n * 4);
  size_t o_xb   = alloc((size_t)n * f_in * 2);
  size_t o_w1t  = alloc((size_t)hid * f_in * 2);
  size_t o_w2t  = alloc((size_t)f_out_pad * hid * 2);
  size_t o_h1   = alloc((size_t)n * hid * 4);          // reused as h2 later
  size_t o_agg1 = alloc((size_t)n * hid * 4);
  size_t o_h1b  = alloc((size_t)n * hid * 2);

  float*          dinv = (float*)(ws + o_dinv);
  unsigned short* xb   = (unsigned short*)(ws + o_xb);
  unsigned short* w1t  = (unsigned short*)(ws + o_w1t);
  unsigned short* w2t  = (unsigned short*)(ws + o_w2t);
  float*          h1   = (float*)(ws + o_h1);
  float*          agg1 = (float*)(ws + o_agg1);
  unsigned short* h1b  = (unsigned short*)(ws + o_h1b);
  float*          h2   = h1;                           // reuse

  const int TB = 256;
  auto gridFor = [&](long long cnt) { return dim3((unsigned)((cnt + TB - 1) / TB)); };

  // degree -> dinv (in place)
  k_fill<<<gridFor(n), TB, 0, stream>>>(dinv, n, 1.0f);            // self loop contributes 1
  k_deg<<<gridFor(E), TB, 0, stream>>>(col, E, dinv);
  k_rsqrt_inplace<<<gridFor(n), TB, 0, stream>>>(dinv, n);

  // layer 1: bf16 conversions + WMMA GEMM
  k_cvt_bf16<<<gridFor((long long)n * f_in), TB, 0, stream>>>(x, xb, (long long)n * f_in);
  k_cvt_transpose_bf16<<<gridFor((long long)hid * f_in), TB, 0, stream>>>(W1, w1t, f_in, hid, hid);
  {
    dim3 g((n + 15) / 16, hid / 256);
    k_gemm_bf16<<<g, 128, 0, stream>>>(xb, w1t, h1, n, f_in, hid);
  }
  k_fill<<<gridFor((long long)n * hid), TB, 0, stream>>>(agg1, (long long)n * hid, 0.0f);
  {
    long long total = E + n;
    k_aggregate<<<dim3((unsigned)((total + 7) / 8)), 256, 0, stream>>>(
        h1, row, col, dinv, E, n, hid, hid, agg1);
  }
  k_bias_relu_cvt<<<gridFor((long long)n * hid), TB, 0, stream>>>(agg1, b1, h1b,
                                                                  (long long)n * hid, hid);

  // layer 2: padded-N WMMA GEMM + aggregation into d_out
  k_cvt_transpose_bf16<<<gridFor((long long)f_out_pad * hid), TB, 0, stream>>>(
      W2, w2t, hid, f_out, f_out_pad);
  {
    dim3 g((n + 15) / 16, f_out_pad / 256);
    k_gemm_bf16<<<g, 128, 0, stream>>>(h1b, w2t, h2, n, hid, f_out_pad);
  }
  k_fill<<<gridFor((long long)n * f_out), TB, 0, stream>>>(out, (long long)n * f_out, 0.0f);
  {
    long long total = E + n;
    k_aggregate<<<dim3((unsigned)((total + 7) / 8)), 256, 0, stream>>>(
        h2, row, col, dinv, E, n, f_out, f_out_pad, out);
  }
  k_bias_relu_f32<<<gridFor((long long)n * f_out), TB, 0, stream>>>(out, b2,
                                                                    (long long)n * f_out, f_out);
}
